// LlamaAttention_82429012345611
// MI455X (gfx1250) — matrix operand order repaired and checked by reading, not compiled, not run on hardware
//
#include <hip/hip_runtime.h>
#include <hip/hip_bf16.h>
#include <stdint.h>

// ---------------------------------------------------------------------------
// Types for CDNA5 WMMA (wave32): v_wmma_f32_16x16x32_bf16
// ---------------------------------------------------------------------------
typedef __attribute__((ext_vector_type(16))) __bf16 v16bf;
typedef __attribute__((ext_vector_type(8)))  float  v8f;
typedef __attribute__((ext_vector_type(8)))  unsigned short u16x8;
typedef __attribute__((ext_vector_type(4)))  unsigned short u16x4;

#define H_DIM 4096
#define N_H   32
#define N_KV  8
#define HD    128
#define SEQ   2048
#define BATCH 2

__device__ __forceinline__ unsigned short f2bf_bits(float f) {
  union { float f; unsigned u; } x; x.f = f;
  unsigned u = x.u;
  return (unsigned short)((u + 0x7FFFu + ((u >> 16) & 1u)) >> 16); // RNE
}
__device__ __forceinline__ float bf2f(unsigned short s) {
  union { unsigned u; float f; } x; x.u = ((unsigned)s) << 16;
  return x.f;
}
__device__ __forceinline__ v16bf make_frag(u16x8 lo, u16x8 hi) {
  union { v16bf v; u16x8 u[2]; } x;
  x.u[0] = lo; x.u[1] = hi;
  return x.v;
}
__device__ __forceinline__ v8f wmma_bf16(v16bf a, v16bf b, v8f c) {
  return __builtin_amdgcn_wmma_f32_16x16x32_bf16(false, a, false, b,
                                                 (short)0, c, false, false);
}
__device__ __forceinline__ void store_out(float* p, float v) { *p = v; }
__device__ __forceinline__ void store_out(unsigned short* p, float v) { *p = f2bf_bits(v); }

// ---------------------------------------------------------------------------
// CDNA5 async global->LDS DMA (ASYNCcnt-tracked). Per lane: 16 bytes from the
// global address to the wave-relative LDS byte address (low 32 bits of a
// generic pointer to __shared__).
// ---------------------------------------------------------------------------
__device__ __forceinline__ void async_b128(void* lds, const void* gptr) {
  asm volatile("global_load_async_to_lds_b128 %0, %1, off"
               :: "v"((unsigned)(uintptr_t)lds), "v"(gptr)
               : "memory");
}
__device__ __forceinline__ void s_wait_async0() {
  asm volatile("s_wait_asynccnt 0x0" ::: "memory");
}

// ---------------------------------------------------------------------------
// Elementwise f32 -> bf16 convert (X activation), 4 elems/thread.
// ---------------------------------------------------------------------------
__global__ void conv_bf16_kernel(const float* __restrict__ X,
                                 unsigned short* __restrict__ Y, int n4)
{
  int i = blockIdx.x * blockDim.x + threadIdx.x;
  if (i >= n4) return;
  float4 v = ((const float4*)X)[i];
  u16x4 o;
  o[0] = f2bf_bits(v.x); o[1] = f2bf_bits(v.y);
  o[2] = f2bf_bits(v.z); o[3] = f2bf_bits(v.w);
  ((u16x4*)Y)[i] = o;
}

// ---------------------------------------------------------------------------
// Weight transpose + convert: W[K,N] f32 -> Wt[N,K] bf16 (LDS tiled, 32x32).
// Block (32,8), grid (N/32, K/32).
// ---------------------------------------------------------------------------
__global__ void transp_bf16_kernel(const float* __restrict__ W,
                                   unsigned short* __restrict__ Wt,
                                   int K, int N)
{
  __shared__ float tile[32][33];
  const int k0 = blockIdx.y * 32;
  const int n0 = blockIdx.x * 32;
  const int tx = threadIdx.x;   // 0..31
  const int ty = threadIdx.y;   // 0..7
  #pragma unroll
  for (int i = 0; i < 4; ++i)
    tile[ty + 8 * i][tx] = W[(size_t)(k0 + ty + 8 * i) * N + n0 + tx];
  __syncthreads();
  #pragma unroll
  for (int i = 0; i < 4; ++i)
    Wt[(size_t)(n0 + ty + 8 * i) * K + k0 + tx] = f2bf_bits(tile[tx][ty + 8 * i]);
}

// ---------------------------------------------------------------------------
// GEMM: C[M,N] = A[M,K](bf16) @ Bt[N,K](bf16)^T + bias[N], f32 accumulate.
// Block = 128 thr (4 waves, 2x2), block tile 128x128, wave tile 64x64, BK=32.
// Async-DMA double-buffered LDS: tile k+1 streams in on ASYNCcnt while tile k
// feeds 16 ds_load_b128 : 16 wmma per wave. One barrier per K-step.
// ---------------------------------------------------------------------------
#define BM 128
#define BN 128
#define BK 32
#define LST 40   // LDS row stride (half-words): mult of 8 -> 16B-aligned frags

__device__ __forceinline__ void gemm_stage_async(
    unsigned short* AsBuf, unsigned short* BsBuf,
    const unsigned short* Ag, const unsigned short* Bg,
    int srow, int scol, int kofs, int K)
{
  #pragma unroll
  for (int p = 0; p < 4; ++p) {
    async_b128(&AsBuf[(p * 32 + srow) * LST + scol], Ag + (size_t)(p * 32) * K + kofs);
    async_b128(&BsBuf[(p * 32 + srow) * LST + scol], Bg + (size_t)(p * 32) * K + kofs);
  }
}

template <typename OutT>
__global__ __launch_bounds__(128) void gemm_kernel(
    const unsigned short* __restrict__ A,
    const unsigned short* __restrict__ Bt,
    const float* __restrict__ bias, OutT* __restrict__ C,
    int M, int N, int K)
{
  __shared__ __align__(16) unsigned short As[2][BM * LST];
  __shared__ __align__(16) unsigned short Bs[2][BN * LST];

  const int tid  = threadIdx.x;
  const int lane = tid & 31;
  const int wave = tid >> 5;
  const int lh   = lane >> 4;
  const int ln   = lane & 15;
  const int waveM = wave & 1;   // 2 M-strips of 64
  const int waveN = wave >> 1;  // 2 N-strips of 64
  const int bm0 = blockIdx.y * BM;
  const int bn0 = blockIdx.x * BN;

  v8f acc[4][4];
  #pragma unroll
  for (int i = 0; i < 4; ++i)
    #pragma unroll
    for (int j = 0; j < 4; ++j)
      acc[i][j] = v8f{0.f,0.f,0.f,0.f,0.f,0.f,0.f,0.f};

  // Staging map: 128x32 bf16 tile; 4 thr/row (16B each), 32 rows/pass, 4 passes.
  const int srow = tid >> 2;        // 0..31
  const int scol = (tid & 3) * 8;   // 0,8,16,24
  const unsigned short* Ag = A  + (size_t)(bm0 + srow) * K + scol;
  const unsigned short* Bg = Bt + (size_t)(bn0 + srow) * K + scol;

  gemm_stage_async(As[0], Bs[0], Ag, Bg, srow, scol, 0, K);
  s_wait_async0();
  __syncthreads();

  int cur = 0;
  for (int k0 = 0; k0 < K; k0 += BK) {
    if (k0 + BK < K)   // DMA next tile while computing this one
      gemm_stage_async(As[cur ^ 1], Bs[cur ^ 1], Ag, Bg, srow, scol, k0 + BK, K);

    const unsigned short* Ab = As[cur];
    const unsigned short* Bb = Bs[cur];
    v16bf af[4], bfr[4];
    #pragma unroll
    for (int i = 0; i < 4; ++i) {
      const unsigned short* p = &Ab[(waveM * 64 + i * 16 + ln) * LST + lh * 8];
      af[i] = make_frag(*(const u16x8*)p, *(const u16x8*)(p + 16));
      const unsigned short* q = &Bb[(waveN * 64 + i * 16 + ln) * LST + lh * 8];
      bfr[i] = make_frag(*(const u16x8*)q, *(const u16x8*)(q + 16));
    }
    #pragma unroll
    for (int i = 0; i < 4; ++i)
      #pragma unroll
      for (int nf = 0; nf < 4; ++nf)
        acc[i][nf] = wmma_bf16(af[i], bfr[nf], acc[i][nf]);

    s_wait_async0();   // own DMAs done; barrier makes all waves' DMAs visible
    __syncthreads();
    cur ^= 1;
  }

  #pragma unroll
  for (int i = 0; i < 4; ++i)
    #pragma unroll
    for (int nf = 0; nf < 4; ++nf) {
      int n = bn0 + waveN * 64 + nf * 16 + ln;
      float bb = bias[n];
      #pragma unroll
      for (int j = 0; j < 8; ++j) {
        int m = bm0 + waveM * 64 + i * 16 + lh * 8 + j;
        store_out(C + (size_t)m * N + n, acc[i][nf][j] + bb);
      }
    }
}

// ---------------------------------------------------------------------------
// RoPE (in place on bf16 buffer laid out (b, s, nh, 128)).
// ---------------------------------------------------------------------------
__global__ void rope_kernel(unsigned short* __restrict__ buf,
                            const int* __restrict__ pos,
                            int nh, int total_i)
{
  int idx = blockIdx.x * blockDim.x + threadIdx.x;
  if (idx >= total_i) return;
  int i  = idx & 63;
  int th = idx >> 6;
  int head = th % nh;
  int tok  = th / nh;
  float p  = (float)pos[tok];
  float inv = __expf(-(float)i * 0.14391156516f); // ln(10000)/64
  float fr = p * inv;
  float s, c;
  __sincosf(fr, &s, &c);
  unsigned short* base = buf + ((size_t)tok * nh + head) * HD;
  float x1 = bf2f(base[i]);
  float x2 = bf2f(base[i + 64]);
  base[i]      = f2bf_bits(x1 * c - x2 * s);
  base[i + 64] = f2bf_bits(x2 * c + x1 * s);
}

// ---------------------------------------------------------------------------
// Flash attention, 64-key tiles. Block = 128 thr (4 waves); each wave owns a
// 16-row Q tile. grid = (SEQ/64, BATCH*N_H). GQA: kv head = h/4.
// K tile staged via async DMA; V tile transposed manually. Output bf16.
// ---------------------------------------------------------------------------
#define KST 136   // 128 + 8 pad (half-words)
#define VST 72    // 64 + 8 pad
#define PST 72

__global__ __launch_bounds__(128) void attn_kernel(
    const unsigned short* __restrict__ Q,
    const unsigned short* __restrict__ K,
    const unsigned short* __restrict__ V,
    const int* __restrict__ mask,
    unsigned short* __restrict__ O)
{
  __shared__ __align__(16) unsigned short Ks[64 * KST];    // K tile row-major
  __shared__ __align__(16) unsigned short Vt[128 * VST];   // V tile transposed
  __shared__ __align__(16) unsigned short Ps[4 * 16 * PST];// per-wave P tiles

  const int tid  = threadIdx.x;
  const int lane = tid & 31;
  const int wave = tid >> 5;
  const int lh   = lane >> 4;
  const int ln   = lane & 15;
  const int bh   = blockIdx.y;
  const int b    = bh >> 5;
  const int h    = bh & 31;
  const int kvh  = h >> 2;
  const int q0   = blockIdx.x * 64 + wave * 16;

  const unsigned short* Qp = Q + ((size_t)b * SEQ) * (N_H * HD) + (size_t)h * HD;
  const unsigned short* Kp = K + ((size_t)b * SEQ) * (N_KV * HD) + (size_t)kvh * HD;
  const unsigned short* Vp = V + ((size_t)b * SEQ) * (N_KV * HD) + (size_t)kvh * HD;
  const int* mb = mask + (size_t)b * SEQ;

  v16bf qa[4];
  #pragma unroll
  for (int c = 0; c < 4; ++c) {
    const unsigned short* p = Qp + (size_t)(q0 + ln) * (N_H * HD) + c * 32 + lh * 8;
    qa[c] = make_frag(*(const u16x8*)p, *(const u16x8*)(p + 16));
  }

  v8f of[8];
  #pragma unroll
  for (int nf = 0; nf < 8; ++nf) of[nf] = v8f{0.f,0.f,0.f,0.f,0.f,0.f,0.f,0.f};
  float mrow[8], lrow[8];
  #pragma unroll
  for (int j = 0; j < 8; ++j) { mrow[j] = -1.0e30f; lrow[j] = 0.f; }

  const float scale = 0.17677669529f;  // 1/sqrt(32)
  const int sr  = tid >> 4;            // 0..7
  const int sc8 = (tid & 15) * 8;      // 0..120

  for (int t = 0; t < SEQ; t += 64) {
    __syncthreads();
    // Stage K tile (64x128) via async DMA; V tile transposed (d x key) manually.
    #pragma unroll
    for (int p4 = 0; p4 < 8; ++p4) {
      int kk = p4 * 8 + sr;
      async_b128(&Ks[kk * KST + sc8], Kp + (size_t)(t + kk) * (N_KV * HD) + sc8);
    }
    #pragma unroll
    for (int p4 = 0; p4 < 8; ++p4) {
      int kk = p4 * 8 + sr;
      u16x8 vv = *(const u16x8*)(Vp + (size_t)(t + kk) * (N_KV * HD) + sc8);
      #pragma unroll
      for (int e = 0; e < 8; ++e) Vt[(sc8 + e) * VST + kk] = vv[e];
    }
    s_wait_async0();
    __syncthreads();

    // S = Q @ K^T (16x64): 4 key-frags x 4 d-chunks = 16 WMMAs.
    v8f sf[4];
    #pragma unroll
    for (int kf = 0; kf < 4; ++kf) sf[kf] = v8f{0.f,0.f,0.f,0.f,0.f,0.f,0.f,0.f};
    #pragma unroll
    for (int c = 0; c < 4; ++c)
      #pragma unroll
      for (int kf = 0; kf < 4; ++kf) {
        const unsigned short* p = &Ks[(kf * 16 + ln) * KST + c * 32 + lh * 8];
        v16bf bk = make_frag(*(const u16x8*)p, *(const u16x8*)(p + 16));
        sf[kf] = wmma_bf16(qa[c], bk, sf[kf]);
      }

    float mbias[4];
    #pragma unroll
    for (int kf = 0; kf < 4; ++kf)
      mbias[kf] = mb[t + kf * 16 + ln] ? 0.f : -3.0e38f;

    // Online softmax: row m = lh*8 + j lives on one 16-lane half.
    unsigned short* Pw = &Ps[wave * 16 * PST];
    #pragma unroll
    for (int j = 0; j < 8; ++j) {
      float sv[4];
      #pragma unroll
      for (int kf = 0; kf < 4; ++kf) sv[kf] = sf[kf][j] * scale + mbias[kf];
      float mx = fmaxf(fmaxf(sv[0], sv[1]), fmaxf(sv[2], sv[3]));
      #pragma unroll
      for (int o = 1; o < 16; o <<= 1) mx = fmaxf(mx, __shfl_xor(mx, o, 16));
      float mnew = fmaxf(mrow[j], mx);
      float corr = __expf(mrow[j] - mnew);
      mrow[j] = mnew;
      float rs = 0.f;
      #pragma unroll
      for (int kf = 0; kf < 4; ++kf) { sv[kf] = __expf(sv[kf] - mnew); rs += sv[kf]; }
      #pragma unroll
      for (int o = 1; o < 16; o <<= 1) rs += __shfl_xor(rs, o, 16);
      lrow[j] = lrow[j] * corr + rs;
      #pragma unroll
      for (int nf = 0; nf < 8; ++nf) of[nf][j] *= corr;
      int m = lh * 8 + j;
      #pragma unroll
      for (int kf = 0; kf < 4; ++kf)
        Pw[m * PST + kf * 16 + ln] = f2bf_bits(sv[kf]);
    }
    __syncthreads();  // P: C-frag layout -> A-frag layout via LDS

    v16bf pa[2];
    #pragma unroll
    for (int c2 = 0; c2 < 2; ++c2) {
      const unsigned short* pp = &Pw[ln * PST + c2 * 32 + lh * 8];
      pa[c2] = make_frag(*(const u16x8*)pp, *(const u16x8*)(pp + 16));
    }
    // O += P @ V : contraction over 64 keys (2 steps) x 8 d-tiles = 16 WMMAs.
    #pragma unroll
    for (int nf = 0; nf < 8; ++nf)
      #pragma unroll
      for (int c2 = 0; c2 < 2; ++c2) {
        const unsigned short* p = &Vt[(nf * 16 + ln) * VST + c2 * 32 + lh * 8];
        v16bf bv = make_frag(*(const u16x8*)p, *(const u16x8*)(p + 16));
        of[nf] = wmma_bf16(pa[c2], bv, of[nf]);
      }
  }

  // Normalize and write O (bf16) at (b, q, h*128 + d).
  #pragma unroll
  for (int j = 0; j < 8; ++j) {
    float inv = 1.0f / lrow[j];
    int m = q0 + lh * 8 + j;
    unsigned short* orow = O + ((size_t)b * SEQ + m) * (size_t)(N_H * HD) + (size_t)h * HD;
    #pragma unroll
    for (int nf = 0; nf < 8; ++nf)
      orow[nf * 16 + ln] = f2bf_bits(of[nf][j] * inv);
  }
}

// ---------------------------------------------------------------------------
// Launch
// ---------------------------------------------------------------------------
extern "C" void kernel_launch(void* const* d_in, const int* in_sizes, int n_in,
                              void* d_out, int out_size, void* d_ws, size_t ws_size,
                              hipStream_t stream)
{
  const float* X    = (const float*)d_in[0];
  const int*   mask = (const int*)d_in[1];
  const int*   pos  = (const int*)d_in[2];
  const float* Wq   = (const float*)d_in[3];
  const float* bq   = (const float*)d_in[4];
  const float* Wk   = (const float*)d_in[5];
  const float* bk   = (const float*)d_in[6];
  const float* Wv   = (const float*)d_in[7];
  const float* bvv  = (const float*)d_in[8];
  const float* Wo   = (const float*)d_in[9];
  const float* bo   = (const float*)d_in[10];

  const int M = BATCH * SEQ;  // 4096 token rows

  char* ws = (char*)d_ws;
  size_t off = 0;
  auto take = [&](size_t bytes) -> char* {
    char* p = ws + off;
    off = (off + bytes + 255) & ~(size_t)255;
    return p;
  };
  unsigned short* Xbf  = (unsigned short*)take((size_t)M * H_DIM * 2);          // 33.6 MB
  unsigned short* WqT  = (unsigned short*)take((size_t)H_DIM * (N_H  * HD) * 2);// 33.6 MB
  unsigned short* WkT  = (unsigned short*)take((size_t)H_DIM * (N_KV * HD) * 2);//  8.4 MB
  unsigned short* WvT  = (unsigned short*)take((size_t)H_DIM * (N_KV * HD) * 2);//  8.4 MB
  unsigned short* WoT  = (unsigned short*)take((size_t)(N_H * HD) * H_DIM * 2); // 33.6 MB
  unsigned short* Qbuf = (unsigned short*)take((size_t)M * (N_H  * HD) * 2);    // 33.6 MB
  unsigned short* Kbuf = (unsigned short*)take((size_t)M * (N_KV * HD) * 2);    //  8.4 MB
  unsigned short* Vbuf = (unsigned short*)take((size_t)M * (N_KV * HD) * 2);    //  8.4 MB
  unsigned short* AObf = (unsigned short*)take((size_t)M * (N_H  * HD) * 2);    // 33.6 MB

  // 1) Precision prep: X -> bf16; weights -> transposed bf16 (L2-resident).
  int n4 = (M * H_DIM) / 4;
  conv_bf16_kernel<<<(n4 + 255) / 256, 256, 0, stream>>>(X, Xbf, n4);
  dim3 tb(32, 8);
  transp_bf16_kernel<<<dim3((N_H  * HD) / 32, H_DIM / 32), tb, 0, stream>>>(Wq, WqT, H_DIM, N_H  * HD);
  transp_bf16_kernel<<<dim3((N_KV * HD) / 32, H_DIM / 32), tb, 0, stream>>>(Wk, WkT, H_DIM, N_KV * HD);
  transp_bf16_kernel<<<dim3((N_KV * HD) / 32, H_DIM / 32), tb, 0, stream>>>(Wv, WvT, H_DIM, N_KV * HD);
  transp_bf16_kernel<<<dim3(H_DIM / 32, (N_H * HD) / 32), tb, 0, stream>>>(Wo, WoT, N_H * HD, H_DIM);

  // 2) QKV projections (bf16 WMMA, bf16 out).
  dim3 gblk(128);
  gemm_kernel<unsigned short><<<dim3((N_H  * HD) / BN, M / BM), gblk, 0, stream>>>(
      Xbf, WqT, bq, Qbuf, M, N_H * HD, H_DIM);
  gemm_kernel<unsigned short><<<dim3((N_KV * HD) / BN, M / BM), gblk, 0, stream>>>(
      Xbf, WkT, bk, Kbuf, M, N_KV * HD, H_DIM);
  gemm_kernel<unsigned short><<<dim3((N_KV * HD) / BN, M / BM), gblk, 0, stream>>>(
      Xbf, WvT, bvv, Vbuf, M, N_KV * HD, H_DIM);

  // 3) RoPE on Q and K.
  int totQ = M * N_H  * 64;
  int totK = M * N_KV * 64;
  rope_kernel<<<(totQ + 255) / 256, 256, 0, stream>>>(Qbuf, pos, N_H,  totQ);
  rope_kernel<<<(totK + 255) / 256, 256, 0, stream>>>(Kbuf, pos, N_KV, totK);

  // 4) Flash attention (bf16 out).
  attn_kernel<<<dim3(SEQ / 64, BATCH * N_H), 128, 0, stream>>>(
      Qbuf, Kbuf, Vbuf, mask, AObf);

  // 5) Output projection (f32 out + bias).
  gemm_kernel<float><<<dim3(H_DIM / BN, M / BM), gblk, 0, stream>>>(
      AObf, WoT, bo, (float*)d_out, M, H_DIM, H_DIM);

  (void)in_sizes; (void)n_in; (void)out_size; (void)ws_size;
}
